// new_MultiHeadAttention_79740362817880
// MI455X (gfx1250) — matrix operand order repaired and checked by reading, not compiled, not run on hardware
//
#include <hip/hip_runtime.h>

// out = (v1+v2) @ (Wo·Wv)^T + (2·Wo·bv + bo)
// (softmax over the size-1 axis is exactly 1, so A/q/k/Wq/Wk are dead code)

typedef _Float16 f16;
typedef __attribute__((ext_vector_type(16))) _Float16 v16h;
typedef __attribute__((ext_vector_type(8)))  _Float16 v8h;
typedef __attribute__((ext_vector_type(8)))  float    v8f;

#define NROWS 16384
#define HID   256
#define KSTEPS 8          // 256 / 32
#define MT_PER_WAVE 8     // M-tiles streamed per wave

#if __has_builtin(__builtin_amdgcn_sched_barrier)
#define SCHED_FENCE() __builtin_amdgcn_sched_barrier(0)
#else
#define SCHED_FENCE()
#endif

// ---------------------------------------------------------------------------
// Kernel 1: W'[i][j] = f16( sum_k Wo[i][k] * Wv[k][j] ),  b'[i] = 2*Wo[i]·bv + bo[i]
// ---------------------------------------------------------------------------
__global__ void fuse_weights(const float* __restrict__ Wo, const float* __restrict__ Wv,
                             const float* __restrict__ bv, const float* __restrict__ bo,
                             f16* __restrict__ Wp, float* __restrict__ bp) {
    const int i = blockIdx.x;
    const int j = threadIdx.x;
    float s = 0.f;
    for (int k = 0; k < HID; ++k)
        s = fmaf(Wo[i * HID + k], Wv[k * HID + j], s);
    Wp[i * HID + j] = (f16)s;
    if (j == 0) {
        float t = 0.f;
        for (int k = 0; k < HID; ++k)
            t = fmaf(Wo[i * HID + k], bv[k], t);
        bp[i] = 2.f * t + bo[i];
    }
}

// ---------------------------------------------------------------------------
// Kernel 2: U = f16(v1 + v2)
// ---------------------------------------------------------------------------
__global__ void sum_to_f16(const float* __restrict__ v1, const float* __restrict__ v2,
                           f16* __restrict__ U) {
    const int idx = blockIdx.x * blockDim.x + threadIdx.x;
    U[idx] = (f16)(v1[idx] + v2[idx]);
}

// ---------------------------------------------------------------------------
// Kernel 3: out[N,256] = U @ Wp^T + bp via v_wmma_f32_16x16x32_f16.
//
// Each wave owns ONE 16-column N-tile; its complete B operand (8 K-steps x
// 8 VGPRs = 64 VGPRs) is register-resident for the whole kernel.
//
// A fragments flow through an 8-slot ring (64 VGPRs), software-pipelined in
// half-tile groups with an enforced schedule:
//     [8 prefetch loads for half-tile H+1]  sched_barrier  [4 WMMAs of H]
// sched_barrier(0) stops the pre-RA scheduler from sinking the loads onto
// their consuming WMMAs (which in rounds 2-3 collapsed the pipeline to one
// 8-VGPR window with s_wait_loadcnt 0 per WMMA). Steady state: 8 b128 loads
// in flight, waits of loadcnt<=8, each load covered by 4 WMMAs + stores.
//
// Grid: (1024/MT_PER_WAVE, 2) blocks x 256 threads (8 waves);
// n_tile = blockIdx.y*8 + wave (16 N-tiles), m-chunk = blockIdx.x.
// ---------------------------------------------------------------------------
__global__ __launch_bounds__(256) void gemm_wmma(const f16* __restrict__ U,
                                                 const f16* __restrict__ Wp,
                                                 const float* __restrict__ bp,
                                                 float* __restrict__ out) {
    const int lane  = threadIdx.x & 31;
    const int wave  = threadIdx.x >> 5;
    const int r     = lane & 15;
    const int hi    = lane >> 4;                 // 0: lanes 0-15, 1: lanes 16-31
    const int col0  = (blockIdx.y * 8 + wave) * 16;
    const int mbase = blockIdx.x * MT_PER_WAVE;  // in M-tile units

    // ---- B operand, register-resident ------------------------------------
    // B[k][n] = Wp[n][k]; B order used here (contiguous k, not ISA table): lane L -> col n = col0 + (L&15),
    // K = k0 + (L>>4)*16 + 0..15, contiguous in memory -> one v16h per K-step.
    const f16* wRow = Wp + (size_t)(col0 + r) * HID + hi * 16;
    v16h bfr[KSTEPS];
#pragma unroll
    for (int ks = 0; ks < KSTEPS; ++ks)
        bfr[ks] = *(const v16h*)(wRow + ks * 32);

    const float bias = bp[col0 + r];

    // ---- A fragments: base + compile-time immediates ----------------------
    // A k order used here (matching B; not the ISA table order): lanes 0-15 hold K {0..15},
    // lanes 16-31 hold K {16..31}. Step it = m*8+ks:
    // offset = m*4096 + ks*32 f16 elements.
    const f16* uBase = U + ((size_t)mbase * 16 + r) * HID + hi * 16;
    auto loadA = [&](int it) -> v16h {
        union { v16h v; v8h h[2]; } a;
        const f16* p = uBase + (it >> 3) * (16 * HID) + (it & 7) * 32;
        a.h[0] = *(const v8h*)(p);        // VGPRs 0-3
        a.h[1] = *(const v8h*)(p + 8);    // VGPRs 4-7
        return a.v;
    };

    // Prologue: fill ring slots 0-3 with half-tile 0 (steps 0..3).
    v16h Ap[8];
#pragma unroll
    for (int j = 0; j < 4; ++j)
        Ap[j] = loadA(j);

    float* oBase = out + ((size_t)mbase * 16 + hi * 8) * HID + col0 + r;

#pragma unroll
    for (int m = 0; m < MT_PER_WAVE; ++m) {
        v8f acc = {};
#pragma unroll
        for (int h = 0; h < 2; ++h) {
            const int base = m * 8 + h * 4;
            // Prefetch the next half-tile into the opposite ring half.
#pragma unroll
            for (int j = 0; j < 4; ++j) {
                const int nit = base + 4 + j;
                if (nit < MT_PER_WAVE * KSTEPS)
                    Ap[nit & 7] = loadA(nit);
            }
            SCHED_FENCE();  // loads above may not sink below; WMMAs may not hoist
#pragma unroll
            for (int j = 0; j < 4; ++j) {
                const int ks = h * 4 + j;   // == (base + j) & 7
                acc = __builtin_amdgcn_wmma_f32_16x16x32_f16(
                    false, Ap[ks], false, bfr[ks], (short)0, acc, false, false);
            }
        }

        // C/D 16x16 f32 layout: VGPR rv -> M = rv (lanes 0-15) / 8+rv (16-31),
        // N = lane & 15. 64B-coalesced per half-wave.
        float* o = oBase + (size_t)m * (16 * HID);
#pragma unroll
        for (int rv = 0; rv < 8; ++rv)
            o[(size_t)rv * HID] = acc[rv] + bias;
    }
}

// ---------------------------------------------------------------------------
// Launch. Inputs (setup_inputs order):
//  0:A 1:q 2:k 3:v1 4:v2 5:Wq 6:bq 7:Wk 8:bk 9:Wv 10:bv 11:Wo 12:bo
// Workspace: [0, 128KB) Wp f16 | [128KB, +1KB) bp f32 | [132096, +8MB) U f16
// ---------------------------------------------------------------------------
extern "C" void kernel_launch(void* const* d_in, const int* in_sizes, int n_in,
                              void* d_out, int out_size, void* d_ws, size_t ws_size,
                              hipStream_t stream) {
    const float* v1 = (const float*)d_in[3];
    const float* v2 = (const float*)d_in[4];
    const float* Wv = (const float*)d_in[9];
    const float* bv = (const float*)d_in[10];
    const float* Wo = (const float*)d_in[11];
    const float* bo = (const float*)d_in[12];
    float* out = (float*)d_out;

    char* ws = (char*)d_ws;
    f16*   Wp = (f16*)(ws);                       // 256*256*2 = 131072 B
    float* bp = (float*)(ws + 131072);            // 1024 B
    f16*   U  = (f16*)(ws + 132096);              // 256B-aligned; 8 MB

    fuse_weights<<<HID, HID, 0, stream>>>(Wo, Wv, bv, bo, Wp, bp);
    sum_to_f16<<<(NROWS * HID) / 256, 256, 0, stream>>>(v1, v2, U);

    dim3 grid((NROWS / 16) / MT_PER_WAVE, 2);
    gemm_wmma<<<grid, 256, 0, stream>>>(U, Wp, bp, out);
}
